// Attention_80152679678101
// MI455X (gfx1250) — matrix operand order repaired and checked by reading, not compiled, not run on hardware
//
#include <hip/hip_runtime.h>
#include <hip/hip_bf16.h>

// ---------------------------------------------------------------------------
// Types
// ---------------------------------------------------------------------------
typedef __bf16 bf16;
typedef __attribute__((ext_vector_type(16))) __bf16 v16bf;
typedef __attribute__((ext_vector_type(8)))  __bf16 bf16x8;
typedef __attribute__((ext_vector_type(8)))  float  v8f;
typedef __attribute__((ext_vector_type(4)))  unsigned int u32x4;
typedef __attribute__((ext_vector_type(8)))  unsigned int u32x8;

#if defined(__has_builtin)
#if __has_builtin(__builtin_amdgcn_s_wait_tensorcnt)
#define WAIT_TENSORCNT(n) __builtin_amdgcn_s_wait_tensorcnt((short)(n))
#endif
#endif
#ifndef WAIT_TENSORCNT
#define WAIT_TENSORCNT(n) asm volatile("s_wait_tensorcnt %0" ::"i"(n) : "memory")
#endif

// ---------------------------------------------------------------------------
// Tensor Data Mover: 2-D tile load Global -> LDS (bf16 elements).
//   lds_off       : byte offset in LDS where the tile lands (row-major, dense)
//   gaddr         : global address of tile start
//   tile_w/tile_h : tile size (elements along contiguous dim, rows)
//   row_stride    : tensor_dim0_stride in elements
// Issued once per wave; tracked with TENSORcnt.
// D# bit packing per CDNA5 ISA ch.8 (group0: count/lds/global/type,
// group1: data_size, dims, tile dims, stride).
// ---------------------------------------------------------------------------
__device__ __forceinline__ void tdm_load_2d(unsigned lds_off, const void* gaddr,
                                            unsigned tile_w, unsigned tile_h,
                                            unsigned long long row_stride) {
  unsigned long long ga = (unsigned long long)gaddr;
  u32x4 g0;
  g0[0] = 1u;                                       // count=1, user mode
  g0[1] = lds_off;                                  // lds_addr
  g0[2] = (unsigned)(ga & 0xFFFFFFFFu);             // global_addr[31:0]
  g0[3] = (unsigned)((ga >> 32) & 0x01FFFFFFu)      // global_addr[56:32]
          | 0x80000000u;                            // type=2 ("image")
  u32x8 g1;
  g1[0] = 0x00010000u;                              // data_size=1 (2 bytes)
  g1[1] = (tile_w & 0xFFFFu) << 16;                 // tensor_dim0[15:0]
  g1[2] = (tile_w >> 16) | ((tile_h & 0xFFFFu) << 16);  // td0 hi | td1 lo
  g1[3] = (tile_h >> 16) | ((tile_w & 0xFFFFu) << 16);  // td1 hi | tile_dim0
  g1[4] = tile_h & 0xFFFFu;                         // tile_dim1 (tile_dim2=0)
  g1[5] = (unsigned)(row_stride & 0xFFFFFFFFu);     // dim0_stride[31:0]
  g1[6] = (unsigned)((row_stride >> 32) & 0xFFFFu); // dim0_stride[47:32]
  g1[7] = 0u;                                       // dim1_stride (unused, 2D)
  asm volatile("tensor_load_to_lds %0, %1" ::"s"(g0), "s"(g1) : "memory");
}

// ---------------------------------------------------------------------------
// WMMA fragment loaders (CDNA5 wave32 layouts, 16x16x32 bf16)
//
// A fragment (16x32, MxK): lane l holds row m = l%16.
//   elems 0..7  -> k = k0 + 8*(l/16) + e        (contiguous 16B)
//   elems 8..15 -> k = k0 + 16 + 8*(l/16) + e   (contiguous 16B)
// B fragment (32x16, KxN): lane l holds col n = l%16.
//   elems 0..7  -> k = k0 + 8*(l/16) + e        (contiguous 16B)
//   elems 8..15 -> k = k0 + 16 + 8*(l/16) + e   (contiguous 16B)
//   B[k][n] == W[n][k] for C = A * W^T, so this reads contiguous W rows.
// C fragment (16x16 f32): lane l, vgpr r -> (m = r + 8*(l/16), n = l%16)
// ---------------------------------------------------------------------------
__device__ __forceinline__ v16bf load_a_frag(const bf16* __restrict__ base,
                                             int ld, int m, int k0, int lane) {
  const bf16* row = base + (size_t)(m + (lane & 15)) * ld + k0 + ((lane >> 4) << 3);
  union { v16bf v; bf16x8 h[2]; } u;
  u.h[0] = *(const bf16x8*)(row);
  u.h[1] = *(const bf16x8*)(row + 16);
  return u.v;
}

__device__ __forceinline__ v16bf load_b_frag(const bf16* __restrict__ base,
                                             int ld, int n, int k0, int lane) {
  const bf16* row = base + (size_t)(n + (lane & 15)) * ld + k0 + ((lane >> 4) << 3);
  union { v16bf v; bf16x8 h[2]; } u;
  u.h[0] = *(const bf16x8*)(row);
  u.h[1] = *(const bf16x8*)(row + 16);
  return u.v;
}

__device__ __forceinline__ v8f wmma_bf16(v16bf a, v16bf b, v8f c) {
  return __builtin_amdgcn_wmma_f32_16x16x32_bf16(
      /*neg_a=*/false, a, /*neg_b=*/false, b,
      /*c_mod=*/(short)0, c, /*reuse_a=*/false, /*reuse_b=*/false);
}

// ---------------------------------------------------------------------------
// fp32 -> bf16 convert
// ---------------------------------------------------------------------------
__global__ void cvt_f32_bf16(const float* __restrict__ in, bf16* __restrict__ out,
                             size_t n) {
  size_t i = (size_t)blockIdx.x * blockDim.x + threadIdx.x;
  size_t stride = (size_t)gridDim.x * blockDim.x;
  for (; i < n; i += stride) out[i] = (bf16)in[i];
}

// ---------------------------------------------------------------------------
// GEMM:  C[m,n] = sum_k A[m,k] * W[n,k]   (A: MxK bf16, W: NxK bf16)
// Block = 256 threads = 8 waves arranged 4(M) x 2(N); block tile 128x128.
// K is consumed in 64-wide steps, double-buffered through LDS via the
// Tensor Data Mover (wave 0 issues, TENSORcnt-waited, barrier-published).
// ---------------------------------------------------------------------------
template <bool F32OUT, bool BIAS>
__global__ __launch_bounds__(256) void gemm_bf16_kernel(
    const bf16* __restrict__ A, const bf16* __restrict__ W,
    const float* __restrict__ bias, float* __restrict__ Cf,
    bf16* __restrict__ Cb, int M, int N, int K) {
  __shared__ __align__(16) bf16 Abuf[2][128][64];
  __shared__ __align__(16) bf16 Wbuf[2][128][64];

  const int lane = threadIdx.x & 31;
  const int w    = threadIdx.x >> 5;
  const int wm   = w & 3;    // 0..3
  const int wn   = w >> 2;   // 0..1
  const int mblk = blockIdx.y * 128;
  const int nblk = blockIdx.x * 128;

  const unsigned aLds[2] = {(unsigned)(uintptr_t)&Abuf[0][0][0],
                            (unsigned)(uintptr_t)&Abuf[1][0][0]};
  const unsigned wLds[2] = {(unsigned)(uintptr_t)&Wbuf[0][0][0],
                            (unsigned)(uintptr_t)&Wbuf[1][0][0]};

  v8f acc[2][4] = {};

  // Prologue: DMA first A/W tiles into buffer 0.
  if (w == 0) {
    tdm_load_2d(aLds[0], A + (size_t)mblk * K, 64, 128, (unsigned long long)K);
    tdm_load_2d(wLds[0], W + (size_t)nblk * K, 64, 128, (unsigned long long)K);
  }

  int buf = 0;
  for (int k0 = 0; k0 < K; k0 += 64) {
    if (w == 0) {
      if (k0 + 64 < K) {
        // Issue next tiles, then wait for the *current* pair (<=2 in flight).
        tdm_load_2d(aLds[buf ^ 1], A + (size_t)mblk * K + (k0 + 64), 64, 128,
                    (unsigned long long)K);
        tdm_load_2d(wLds[buf ^ 1], W + (size_t)nblk * K + (k0 + 64), 64, 128,
                    (unsigned long long)K);
        WAIT_TENSORCNT(2);
      } else {
        WAIT_TENSORCNT(0);
      }
    }
    __syncthreads();  // current buffer is valid for everyone

    const bf16* at = &Abuf[buf][0][0];
    const bf16* wt = &Wbuf[buf][0][0];
#pragma unroll
    for (int kk = 0; kk < 2; ++kk) {
      v16bf afrag[2], bfrag[4];
#pragma unroll
      for (int i = 0; i < 2; ++i)
        afrag[i] = load_a_frag(at, 64, wm * 32 + 16 * i, kk * 32, lane);
#pragma unroll
      for (int j = 0; j < 4; ++j)
        bfrag[j] = load_b_frag(wt, 64, wn * 64 + 16 * j, kk * 32, lane);
#pragma unroll
      for (int i = 0; i < 2; ++i)
#pragma unroll
        for (int j = 0; j < 4; ++j)
          acc[i][j] = wmma_bf16(afrag[i], bfrag[j], acc[i][j]);
    }

    asm volatile("s_wait_dscnt 0" ::: "memory");  // LDS reads landed
    __syncthreads();  // safe for next TDM to overwrite buf^1
    buf ^= 1;
  }

  const int nloc = lane & 15;
  const int mhi  = (lane >> 4) * 8;
#pragma unroll
  for (int i = 0; i < 2; ++i)
#pragma unroll
    for (int j = 0; j < 4; ++j) {
      const int col = nblk + wn * 64 + 16 * j + nloc;
      float bv = 0.0f;
      if (BIAS) bv = bias[col];
#pragma unroll
      for (int r = 0; r < 8; ++r) {
        const int rowm = mblk + wm * 32 + 16 * i + mhi + r;
        const float v = acc[i][j][r] + bv;
        if (F32OUT) Cf[(size_t)rowm * N + col] = v;
        else        Cb[(size_t)rowm * N + col] = (bf16)v;
      }
    }
}

// ---------------------------------------------------------------------------
// RoPE + head scatter.
// qkv: (B, N, 3D) bf16.  Writes per-head:
//   Qh: (B*H, N, 128) with rope applied and folded 1/sqrt(128) scale
//   Kh: (B*H, N, 128) with rope applied
//   VT: (B*H, 128, N)  (transposed V for direct B-fragment loads)
// One thread per (b, h, n, i), i in 0..63 (a rope pair + two V elements).
// ---------------------------------------------------------------------------
__global__ void rope_scatter_kernel(const bf16* __restrict__ qkv,
                                    bf16* __restrict__ Qh,
                                    bf16* __restrict__ Kh,
                                    bf16* __restrict__ VT) {
  const int Nn = 2048, Hh = 16;
  size_t tid = (size_t)blockIdx.x * blockDim.x + threadIdx.x;
  const int i = (int)(tid & 63);
  size_t t = tid >> 6;
  const int n = (int)(t % Nn); t /= Nn;
  const int h = (int)(t % Hh);
  const int b = (int)(t / Hh);

  const bf16* row = qkv + ((size_t)(b * Nn + n)) * 6144;
  const float q1 = (float)row[h * 128 + i];
  const float q2 = (float)row[h * 128 + i + 64];
  const float k1 = (float)row[2048 + h * 128 + i];
  const float k2 = (float)row[2048 + h * 128 + i + 64];

  // inv_freq = 10000^{-(2i)/128} = exp(-i * (2/128) * ln(10000))
  const float inv = __expf((float)i * (-2.0f / 128.0f) * 9.210340371976184f);
  const float ang = (float)n * inv;
  float s, c;
  __sincosf(ang, &s, &c);
  const float qscale = 0.08838834764831845f;  // 1/sqrt(128)

  const size_t base = ((size_t)(b * Hh + h)) * Nn * 128 + (size_t)n * 128;
  Qh[base + i]      = (bf16)((q1 * c - q2 * s) * qscale);
  Qh[base + i + 64] = (bf16)((q1 * s + q2 * c) * qscale);
  Kh[base + i]      = (bf16)(k1 * c - k2 * s);
  Kh[base + i + 64] = (bf16)(k1 * s + k2 * c);

  const size_t vbase = ((size_t)(b * Hh + h)) * 128 * Nn;
  VT[vbase + (size_t)i * Nn + n]        = row[4096 + h * 128 + i];
  VT[vbase + (size_t)(i + 64) * Nn + n] = row[4096 + h * 128 + i + 64];
}

// ---------------------------------------------------------------------------
// Flash attention.  Grid (N/128, H, B), block 256 = 8 waves.
// Each wave owns one 16-row Q tile; all waves walk the same 64-wide KV
// blocks, which are staged K:(64x128) / V^T:(128x64) into LDS by the Tensor
// Data Mover, double-buffered (wave 0 issues, s_wait_tensorcnt, barrier).
// S = Q*K^T via WMMA; online softmax in f32; P staged through per-wave LDS
// (C-layout -> A-layout cross-lane shuffle); out += P*V via WMMA.
// ---------------------------------------------------------------------------
__global__ __launch_bounds__(256) void flash_attn_kernel(
    const bf16* __restrict__ Qh, const bf16* __restrict__ Kh,
    const bf16* __restrict__ VT, bf16* __restrict__ O) {
  __shared__ __align__(16) bf16 Kbuf[2][64][128];   // (keypos, hd)
  __shared__ __align__(16) bf16 Vbuf[2][128][64];   // (hd, keypos)
  __shared__ __align__(16) bf16 Plds[8][16][64];

  const int Nn = 2048, HD = 128;
  const int lane = threadIdx.x & 31;
  const int w    = threadIdx.x >> 5;
  const int h = blockIdx.y;
  const int b = blockIdx.z;
  const int qbase = blockIdx.x * 128 + w * 16;

  const size_t headoff = ((size_t)(b * 16 + h)) * Nn * HD;
  const bf16* q  = Qh + headoff;
  const bf16* kp = Kh + headoff;
  const bf16* vt = VT + headoff;  // (HD, Nn)

  const unsigned kLds[2] = {(unsigned)(uintptr_t)&Kbuf[0][0][0],
                            (unsigned)(uintptr_t)&Kbuf[1][0][0]};
  const unsigned vLds[2] = {(unsigned)(uintptr_t)&Vbuf[0][0][0],
                            (unsigned)(uintptr_t)&Vbuf[1][0][0]};

  // Q fragments for the whole head dim (4 x 32-k chunks)
  v16bf qf[4];
#pragma unroll
  for (int c = 0; c < 4; ++c) qf[c] = load_a_frag(q, HD, qbase, 32 * c, lane);

  v8f oacc[8] = {};
  float mrow[8], lrow[8];
#pragma unroll
  for (int r = 0; r < 8; ++r) { mrow[r] = -1e30f; lrow[r] = 0.0f; }

  // Prologue: DMA first K/V tiles into buffer 0.
  if (w == 0) {
    tdm_load_2d(kLds[0], kp, 128, 64, (unsigned long long)HD);
    tdm_load_2d(vLds[0], vt, 64, 128, (unsigned long long)Nn);
  }

  int buf = 0;
  for (int j0 = 0; j0 < Nn; j0 += 64) {
    if (w == 0) {
      if (j0 + 64 < Nn) {
        tdm_load_2d(kLds[buf ^ 1], kp + (size_t)(j0 + 64) * HD, 128, 64,
                    (unsigned long long)HD);
        tdm_load_2d(vLds[buf ^ 1], vt + (j0 + 64), 64, 128,
                    (unsigned long long)Nn);
        WAIT_TENSORCNT(2);
      } else {
        WAIT_TENSORCNT(0);
      }
    }
    __syncthreads();  // current K/V tiles valid for everyone

    const bf16* kt = &Kbuf[buf][0][0];
    const bf16* vtile = &Vbuf[buf][0][0];

    // ---- scores S (16 x 64), already scaled via Q prescale ----
    v8f s[4] = {};
#pragma unroll
    for (int t = 0; t < 4; ++t) {
#pragma unroll
      for (int c = 0; c < 4; ++c) {
        v16bf kf = load_b_frag(kt, HD, 16 * t, 32 * c, lane);
        s[t] = wmma_bf16(qf[c], kf, s[t]);
      }
    }

    // ---- row max (partial over 4 tiles, then across the 16-lane half) ----
    float pm[8];
#pragma unroll
    for (int r = 0; r < 8; ++r) {
      float v = s[0][r];
#pragma unroll
      for (int t = 1; t < 4; ++t) v = fmaxf(v, s[t][r]);
      pm[r] = v;
    }
#pragma unroll
    for (int off = 1; off < 16; off <<= 1)
#pragma unroll
      for (int r = 0; r < 8; ++r)
        pm[r] = fmaxf(pm[r], __shfl_xor(pm[r], off, 16));

    float corr[8];
#pragma unroll
    for (int r = 0; r < 8; ++r) {
      const float mnew = fmaxf(mrow[r], pm[r]);
      corr[r] = __expf(mrow[r] - mnew);
      mrow[r] = mnew;
    }
#pragma unroll
    for (int d = 0; d < 8; ++d)
#pragma unroll
      for (int r = 0; r < 8; ++r) oacc[d][r] *= corr[r];

    // ---- P = exp(S - m), row sums, and P -> LDS (bf16) ----
    float psum[8];
#pragma unroll
    for (int r = 0; r < 8; ++r) psum[r] = 0.0f;
    const int mloc = (lane >> 4) * 8;
    const int nloc = lane & 15;
#pragma unroll
    for (int t = 0; t < 4; ++t)
#pragma unroll
      for (int r = 0; r < 8; ++r) {
        const float p = __expf(s[t][r] - mrow[r]);
        psum[r] += p;
        Plds[w][mloc + r][t * 16 + nloc] = (bf16)p;
      }
#pragma unroll
    for (int off = 1; off < 16; off <<= 1)
#pragma unroll
      for (int r = 0; r < 8; ++r) psum[r] += __shfl_xor(psum[r], off, 16);
#pragma unroll
    for (int r = 0; r < 8; ++r) lrow[r] = lrow[r] * corr[r] + psum[r];

    asm volatile("s_wait_dscnt 0" ::: "memory");

    // ---- re-load P as A-fragments (16 x 64 -> 2 chunks of 32k) ----
    v16bf pf[2];
#pragma unroll
    for (int c = 0; c < 2; ++c) {
      const bf16* prow = &Plds[w][lane & 15][32 * c + ((lane >> 4) << 3)];
      union { v16bf v; bf16x8 hh[2]; } u;
      u.hh[0] = *(const bf16x8*)(prow);
      u.hh[1] = *(const bf16x8*)(prow + 16);
      pf[c] = u.v;
    }

    // ---- out += P * V  (V fragments from the staged V^T tile) ----
#pragma unroll
    for (int d = 0; d < 8; ++d)
#pragma unroll
      for (int c = 0; c < 2; ++c) {
        v16bf vf = load_b_frag(vtile, 64, d * 16, 32 * c, lane);
        oacc[d] = wmma_bf16(pf[c], vf, oacc[d]);
      }

    asm volatile("s_wait_dscnt 0" ::: "memory");  // LDS reads landed
    __syncthreads();  // safe for next TDM to overwrite buf^1
    buf ^= 1;
  }

  // ---- epilogue: normalize and scatter to (B, N, D) bf16 ----
  const int nloc = lane & 15;
  const int mhi  = (lane >> 4) * 8;
#pragma unroll
  for (int r = 0; r < 8; ++r) {
    const float inv = 1.0f / lrow[r];
    const int qrow = qbase + mhi + r;
    const size_t obase = ((size_t)(b * Nn) + qrow) * 2048 + (size_t)h * 128;
#pragma unroll
    for (int d = 0; d < 8; ++d)
      O[obase + d * 16 + nloc] = (bf16)(oacc[d][r] * inv);
  }
}

// ---------------------------------------------------------------------------
// Host-side launcher
// ---------------------------------------------------------------------------
extern "C" void kernel_launch(void* const* d_in, const int* in_sizes, int n_in,
                              void* d_out, int out_size, void* d_ws,
                              size_t ws_size, hipStream_t stream) {
  (void)in_sizes; (void)n_in; (void)out_size; (void)ws_size;
  const float* x      = (const float*)d_in[0];  // (2, 2048, 2048)
  const float* w_qkv  = (const float*)d_in[1];  // (6144, 2048)
  const float* w_proj = (const float*)d_in[2];  // (2048, 2048)
  const float* b_proj = (const float*)d_in[3];  // (2048,)
  float* out = (float*)d_out;                   // (2, 2048, 2048)

  const int B = 2, Nn = 2048, D = 2048, H = 16, HD = 128;
  const size_t M = (size_t)B * Nn;  // 4096

  char* ws = (char*)d_ws;
  size_t off = 0;
  auto alloc = [&](size_t bytes) -> char* {
    char* p = ws + off;
    off = (off + bytes + 255) & ~(size_t)255;
    return p;
  };

  bf16* xb     = (bf16*)alloc(M * D * sizeof(bf16));  // reused as attn out
  bf16* wqkvb  = (bf16*)alloc((size_t)3 * D * D * sizeof(bf16));
  bf16* wprojb = (bf16*)alloc((size_t)D * D * sizeof(bf16));
  bf16* qkvb   = (bf16*)alloc(M * 3 * D * sizeof(bf16));
  bf16* Qhb    = (bf16*)alloc((size_t)B * H * Nn * HD * sizeof(bf16));
  bf16* Khb    = (bf16*)alloc((size_t)B * H * Nn * HD * sizeof(bf16));
  bf16* VTb    = (bf16*)alloc((size_t)B * H * HD * Nn * sizeof(bf16));
  bf16* attnO  = xb;  // x dead after QKV GEMM; reuse its slot

  // 1) fp32 -> bf16 conversions
  cvt_f32_bf16<<<4096, 256, 0, stream>>>(x, xb, M * D);
  cvt_f32_bf16<<<4096, 256, 0, stream>>>(w_qkv, wqkvb, (size_t)3 * D * D);
  cvt_f32_bf16<<<4096, 256, 0, stream>>>(w_proj, wprojb, (size_t)D * D);

  // 2) qkv = x @ w_qkv^T     (M=4096, N=6144, K=2048), bf16 out
  {
    dim3 grid(3 * D / 128, (int)(M / 128));
    gemm_bf16_kernel<false, false><<<grid, 256, 0, stream>>>(
        xb, wqkvb, nullptr, nullptr, qkvb, (int)M, 3 * D, D);
  }

  // 3) RoPE + per-head scatter (+ V transpose)
  {
    const size_t total = (size_t)B * H * Nn * 64;  // 4.19M threads
    rope_scatter_kernel<<<(unsigned)(total / 256), 256, 0, stream>>>(
        qkvb, Qhb, Khb, VTb);
  }

  // 4) flash attention -> attnO (B, N, D) bf16
  {
    dim3 grid(Nn / 128, H, B);
    flash_attn_kernel<<<grid, 256, 0, stream>>>(Qhb, Khb, VTb, attnO);
  }

  // 5) out = attnO @ w_proj^T + b_proj   (M=4096, N=2048, K=2048), fp32 out
  {
    dim3 grid(D / 128, (int)(M / 128));
    gemm_bf16_kernel<true, true><<<grid, 256, 0, stream>>>(
        attnO, wprojb, b_proj, out, nullptr, (int)M, D, D);
  }
}
